// RecurrentEncoder_17343077941611
// MI455X (gfx1250) — matrix operand order repaired and checked by reading, not compiled, not run on hardware
//
#include <hip/hip_runtime.h>

// Problem constants (match reference)
#define B_  32
#define T_  1024
#define H_  768
#define IN_ 1536   // 2*H
#define L_  2
#define G3_ 2304   // 3*H
#define KI_ (H_ / 32)   // 24 k-iterations in the recurrent GEMM

typedef __attribute__((ext_vector_type(16))) __bf16 v16bf;
typedef __attribute__((ext_vector_type(8)))  float  v8f;

union Frag16 { v16bf v; int4 q[2]; };

// ---------------------------------------------------------------------------
// WMMA fragment loaders (wave32), contiguous k order used for both A and B:
//  A 16x32 bf16: lane L (r=L&15, s=L>>4) holds row M=r,
//                elems 0..15 = K [16s .. 16s+15]   (32B contiguous)
//  B 32x16 bf16: lane L (c=L&15, s=L>>4) holds col N=c,
//                elems 0..15 = K [16s .. 16s+15]   (32B contiguous)
//  C/D 16x16 f32: lane L elem j -> (M = j + 8s, N = c)
// ---------------------------------------------------------------------------
__device__ __forceinline__ v16bf load_a_frag(const __bf16* A, int lda,
                                             int mBase, int k0, int lane) {
  const int r = lane & 15, s = lane >> 4;
  const __bf16* p = A + (size_t)(mBase + r) * lda + k0 + 16 * s;
  Frag16 f;
  f.q[0] = *reinterpret_cast<const int4*>(p);
  f.q[1] = *reinterpret_cast<const int4*>(p + 8);
  return f.v;
}

__device__ __forceinline__ v16bf load_b_frag(const __bf16* W, int ldb,
                                             int nBase, int k0, int lane) {
  const int c = lane & 15, s = lane >> 4;
  const __bf16* p = W + (size_t)(nBase + c) * ldb + k0 + 16 * s;
  Frag16 f;
  f.q[0] = *reinterpret_cast<const int4*>(p);
  f.q[1] = *reinterpret_cast<const int4*>(p + 8);
  return f.v;
}

// ---------------------------------------------------------------------------
// fp32 -> bf16 conversion (grid-stride)
// ---------------------------------------------------------------------------
__global__ void f32_to_bf16_kernel(const float* __restrict__ in,
                                   __bf16* __restrict__ out, long n) {
  long i = (long)blockIdx.x * blockDim.x + threadIdx.x;
  const long stride = (long)gridDim.x * blockDim.x;
  for (; i < n; i += stride) out[i] = (__bf16)in[i];
}

// ---------------------------------------------------------------------------
// xg = In[B*T, IN] x Wih[dir][3H, IN]^T + bih
// Register-blocked: each wave computes a 64x32 macro-tile (4 M x 2 N WMMA
// tiles): 8 WMMA per 6 fragment loads -> 1.5 b128/WMMA instead of 4.
// grid = ((B*T/64)*(3H/32)/8, 2), block = 256 (8 waves)
// ---------------------------------------------------------------------------
__global__ __launch_bounds__(256)
void gemm_xg_kernel(const __bf16* __restrict__ In,   // [B*T, IN] bf16
                    const __bf16* __restrict__ Wih,  // [2][3H, IN] bf16 (layer)
                    const float*  __restrict__ bih,  // [2][3H]
                    float*        __restrict__ xg)   // [2][B*T, 3H]
{
  const int  lane = threadIdx.x & 31;
  const long job  = (long)blockIdx.x * (blockDim.x >> 5) + (threadIdx.x >> 5);
  const int  dir  = blockIdx.y;
  const int  NP   = G3_ / 32;                  // 72 N-pairs
  const long MQ   = (long)B_ * T_ / 64;        // 512 M-quads
  if (job >= MQ * NP) return;
  const int np = (int)(job % NP);
  const int mq = (int)(job / NP);
  const int mBase = mq * 64, nBase = np * 32;
  const __bf16* W = Wih + (size_t)dir * G3_ * IN_;

  v8f c[8] = {};                               // c[mi*2+nj]
  for (int k0 = 0; k0 < IN_; k0 += 32) {
    v16bf a[4], b[2];
#pragma unroll
    for (int i = 0; i < 4; ++i) a[i] = load_a_frag(In, IN_, mBase + 16 * i, k0, lane);
#pragma unroll
    for (int j = 0; j < 2; ++j) b[j] = load_b_frag(W, IN_, nBase + 16 * j, k0, lane);
#pragma unroll
    for (int i = 0; i < 4; ++i)
#pragma unroll
      for (int j = 0; j < 2; ++j)
        c[i * 2 + j] = __builtin_amdgcn_wmma_f32_16x16x32_bf16(
            false, a[i], false, b[j], (short)0, c[i * 2 + j], false, false);
  }

  float* outB = xg + (size_t)dir * B_ * T_ * G3_;
  const int s = lane >> 4;
#pragma unroll
  for (int j = 0; j < 2; ++j) {
    const int   col  = nBase + 16 * j + (lane & 15);
    const float bias = bih[(size_t)dir * G3_ + col];
#pragma unroll
    for (int i = 0; i < 4; ++i) {
      const int mRow = mBase + 16 * i;
#pragma unroll
      for (int e = 0; e < 8; ++e)
        outB[(size_t)(mRow + e + 8 * s) * G3_ + col] = c[i * 2 + j][e] + bias;
    }
  }
}

// ---------------------------------------------------------------------------
// software grid barrier (cumulative counter + per-block generation)
// ---------------------------------------------------------------------------
__device__ __forceinline__ void gsync(unsigned* cnt, unsigned* gen, unsigned nb) {
  __threadfence();
  __syncthreads();
  if (threadIdx.x == 0) {
    const unsigned target = (++(*gen)) * nb;
    atomicAdd(cnt, 1u);
    volatile unsigned* vc = cnt;
    while (*vc < target) __builtin_amdgcn_s_sleep(1);
  }
  __syncthreads();
  __threadfence();
}

// ---------------------------------------------------------------------------
// Persistent bidirectional GRU scan over T for one layer.
// grid = 72 blocks x 128 thr = 288 waves = 2 dirs x 144 N-tiles.
// Each wave owns one N-tile of Whh: its 24 B-fragments (24 KB) are staged
// into LDS ONCE, then all 1024 steps read them via ds_load_b128 -- Whh
// leaves L2 once per layer instead of once per timestep.
// Per step: phase1 gh = hB x Whh^T (2 WMMA per k-iter, B-frag reused);
// barrier; phase2 gate math; barrier.
// ---------------------------------------------------------------------------
__global__ __launch_bounds__(128)
void gru_scan_kernel(const float*  __restrict__ xg,     // [2][B*T, 3H] (incl. bih)
                     const __bf16* __restrict__ Whh,    // [2][3H, H] bf16
                     const float*  __restrict__ bhh,    // [2][3H]
                     float*        __restrict__ gh,     // [2][B, 3H] scratch
                     float*        __restrict__ hF,     // [2][B, H] fp32 h (zeroed)
                     __bf16*       __restrict__ hB,     // [2][B, H] bf16 h (zeroed)
                     __bf16*       __restrict__ Ynext,  // [B*T, 2H] next-layer in
                     float*        __restrict__ outX,   // [B*T, 2H] final output
                     float*        __restrict__ outHid, // [2][B, H] h_last slot
                     unsigned*     __restrict__ barCnt, // zeroed
                     int lastLayer)
{
  // [waveInBlk][kIter][lane][16 bf16] = 4*24*32*16*2B = 96 KB (gfx1250: 320 KB)
  __shared__ __bf16 WhhLds[4][KI_][32][16];

  __shared__ unsigned gen;
  if (threadIdx.x == 0) gen = 0;

  const unsigned nb  = gridDim.x;
  const int lane     = threadIdx.x & 31;
  const int w        = threadIdx.x >> 5;                          // wave in block
  const int wave     = blockIdx.x * (blockDim.x >> 5) + w;        // 0..287
  const int tid      = blockIdx.x * blockDim.x + threadIdx.x;
  const int nthreads = gridDim.x * blockDim.x;
  const int dirW     = wave / 144;
  const int ntW      = wave % 144;

  // ---- stage this wave's Whh B-fragments into LDS (once per layer) ----
  {
    const __bf16* W = Whh + (size_t)dirW * G3_ * H_;
    for (int kk = 0; kk < KI_; ++kk) {
      Frag16 f;
      f.v = load_b_frag(W, H_, ntW * 16, kk * 32, lane);
      *reinterpret_cast<int4*>(&WhhLds[w][kk][lane][0]) = f.q[0];
      *reinterpret_cast<int4*>(&WhhLds[w][kk][lane][8]) = f.q[1];
    }
  }
  __syncthreads();

  for (int t = 0; t < T_; ++t) {
    // ---- phase 1: gh[dir] = hB[dir] x Whh[dir]^T (LDS-resident B) ----
    {
      const __bf16* A = hB + (size_t)dirW * B_ * H_;
      v8f c0 = {}, c1 = {};
      for (int kk = 0; kk < KI_; ++kk) {
        const int k0 = kk * 32;
        v16bf a0 = load_a_frag(A, H_, 0,  k0, lane);
        v16bf a1 = load_a_frag(A, H_, 16, k0, lane);
        Frag16 fb;
        fb.q[0] = *reinterpret_cast<const int4*>(&WhhLds[w][kk][lane][0]);
        fb.q[1] = *reinterpret_cast<const int4*>(&WhhLds[w][kk][lane][8]);
        c0 = __builtin_amdgcn_wmma_f32_16x16x32_bf16(false, a0, false, fb.v,
                                                     (short)0, c0, false, false);
        c1 = __builtin_amdgcn_wmma_f32_16x16x32_bf16(false, a1, false, fb.v,
                                                     (short)0, c1, false, false);
      }
      float* outG = gh + (size_t)dirW * B_ * G3_;
      const int col = ntW * 16 + (lane & 15), s = lane >> 4;
#pragma unroll
      for (int e = 0; e < 8; ++e) {
        outG[(size_t)(e + 8 * s) * G3_ + col]      = c0[e];
        outG[(size_t)(16 + e + 8 * s) * G3_ + col] = c1[e];
      }
    }
    gsync(barCnt, &gen, nb);

    // ---- phase 2: gates + h update ----
    for (int idx = tid; idx < 2 * B_ * H_; idx += nthreads) {
      const int dir = idx / (B_ * H_);
      const int rem = idx % (B_ * H_);
      const int b   = rem / H_, n = rem % H_;
      const int ta  = dir ? (T_ - 1 - t) : t;

      const float* xrow = xg + ((size_t)dir * B_ * T_ + (size_t)b * T_ + ta) * G3_;
      const float* grow = gh + ((size_t)dir * B_ + b) * G3_;
      const float* bh   = bhh + (size_t)dir * G3_;

      const float xr = xrow[n], xz = xrow[H_ + n], xn = xrow[2 * H_ + n];
      const float hr = grow[n]          + bh[n];
      const float hz = grow[H_ + n]     + bh[H_ + n];
      const float hn = grow[2 * H_ + n] + bh[2 * H_ + n];

      const float r  = 1.f / (1.f + __expf(-(xr + hr)));
      const float z  = 1.f / (1.f + __expf(-(xz + hz)));
      const float nn = tanhf(xn + r * hn);

      const size_t hoff  = (size_t)dir * B_ * H_ + (size_t)b * H_ + n;
      const float  hprev = hF[hoff];
      const float  hnew  = (1.f - z) * nn + z * hprev;
      hF[hoff] = hnew;
      hB[hoff] = (__bf16)hnew;

      const size_t yoff = ((size_t)b * T_ + ta) * (2 * H_) + (size_t)dir * H_ + n;
      Ynext[yoff] = (__bf16)hnew;
      if (lastLayer)   outX[yoff]   = hnew;
      if (t == T_ - 1) outHid[hoff] = hnew;   // h_last (fwd: t=T-1; bwd: t=0)
    }
    gsync(barCnt, &gen, nb);
  }
}

// ---------------------------------------------------------------------------
extern "C" void kernel_launch(void* const* d_in, const int* in_sizes, int n_in,
                              void* d_out, int out_size, void* d_ws, size_t ws_size,
                              hipStream_t stream) {
  (void)in_sizes; (void)n_in; (void)out_size; (void)ws_size;
  const float* X   = (const float*)d_in[0];
  // d_in[1] = len_X (unused, use_packed=False)
  const float* Wih = (const float*)d_in[2];  // [L,2,3H,IN]
  const float* Whh = (const float*)d_in[3];  // [L,2,3H,H]
  const float* bih = (const float*)d_in[4];  // [L,2,3H]
  const float* bhh = (const float*)d_in[5];  // [L,2,3H]
  float* out = (float*)d_out;                // [B,T,2H] ++ [L*2,B,H]

  char* ws = (char*)d_ws;
  size_t off = 0;
  auto take = [&](size_t bytes) -> char* {
    char* p = ws + off;
    off = (off + bytes + 255) & ~(size_t)255;
    return p;
  };
  __bf16*   Xbf  = (__bf16*)take((size_t)B_ * T_ * IN_ * 2);       // ~100.7 MB
  __bf16*   Ybf  = (__bf16*)take((size_t)B_ * T_ * IN_ * 2);       // ~100.7 MB
  __bf16*   WihB = (__bf16*)take((size_t)L_ * 2 * G3_ * IN_ * 2);  // ~28.3 MB
  __bf16*   WhhB = (__bf16*)take((size_t)L_ * 2 * G3_ * H_ * 2);   // ~14.2 MB
  float*    xg   = (float*) take((size_t)2 * B_ * T_ * G3_ * 4);   // ~604 MB
  float*    gh   = (float*) take((size_t)2 * B_ * G3_ * 4);
  float*    hF   = (float*) take((size_t)2 * B_ * H_ * 4);
  __bf16*   hB   = (__bf16*)take((size_t)2 * B_ * H_ * 2);
  unsigned* bar  = (unsigned*)take(256);

  // one-time bf16 conversions
  f32_to_bf16_kernel<<<4096, 256, 0, stream>>>(X,   Xbf,  (long)B_ * T_ * IN_);
  f32_to_bf16_kernel<<<2048, 256, 0, stream>>>(Wih, WihB, (long)L_ * 2 * G3_ * IN_);
  f32_to_bf16_kernel<<<2048, 256, 0, stream>>>(Whh, WhhB, (long)L_ * 2 * G3_ * H_);

  const int jobsPerDir = (B_ * T_ / 64) * (G3_ / 32);  // 512*72 = 36864 wave jobs
  const int gemmBlocks = jobsPerDir / 8;               // 4608 (exact)

  for (int l = 0; l < L_; ++l) {
    const __bf16* In    = (l == 0) ? Xbf : Ybf;
    __bf16*       Ynext = (l == L_ - 1) ? Xbf : Ybf;   // last layer: dead sink

    gemm_xg_kernel<<<dim3(gemmBlocks, 2), 256, 0, stream>>>(
        In, WihB + (size_t)l * 2 * G3_ * IN_, bih + (size_t)l * 2 * G3_, xg);

    hipMemsetAsync(hF,  0, (size_t)2 * B_ * H_ * 4, stream);
    hipMemsetAsync(hB,  0, (size_t)2 * B_ * H_ * 2, stream);
    hipMemsetAsync(bar, 0, 256, stream);

    gru_scan_kernel<<<72, 128, 0, stream>>>(
        xg, WhhB + (size_t)l * 2 * G3_ * H_, bhh + (size_t)l * 2 * G3_,
        gh, hF, hB, Ynext, out,
        out + (size_t)B_ * T_ * 2 * H_ + (size_t)l * 2 * B_ * H_,
        bar, (l == L_ - 1) ? 1 : 0);
  }
}
